// GraspSampler_4002909519891
// MI455X (gfx1250) — matrix operand order repaired and checked by reading, not compiled, not run on hardware
//
#include <hip/hip_runtime.h>
#include <hip/hip_bf16.h>

// ---------------------------------------------------------------------------
// CDNA5 (gfx1250) fused PointNet++ grasp sampler.
// wave32; matrix ops via v_wmma_f32_16x16x32_f16. SA blocks fully fused in LDS
// (gather -> 3x WMMA MLP -> maxpool) so only pooled features hit HBM.
// Weights are pre-swizzled into WMMA B-fragment order so every B operand is a
// single contiguous 32-byte load per lane.
// ---------------------------------------------------------------------------

typedef __attribute__((ext_vector_type(16))) _Float16 v16h;
typedef __attribute__((ext_vector_type(8)))  _Float16 v8h;
typedef __attribute__((ext_vector_type(8)))  float    v8f;

// ---- WMMA fragment loaders (layouts per CDNA5 ISA 7.12.2, wave32) ----------
// A: 16x32 f16. lane L<16 holds row M=L, K = {0..7, 16..23}; lane L>=16 holds
//    row M=L-16, K = {8..15, 24..31}. Two 16B vector loads per lane.
__device__ __forceinline__ v16h wmma_load_a(const _Float16* X, int ld, int m0,
                                            int k0, int lane) {
  const _Float16* p = X + (m0 + (lane & 15)) * ld + k0 + ((lane & 16) ? 8 : 0);
  const v8h lo = *(const v8h*)p;
  const v8h hi = *(const v8h*)(p + 16);
  return __builtin_shufflevector(lo, hi, 0, 1, 2, 3, 4, 5, 6, 7,
                                 8, 9, 10, 11, 12, 13, 14, 15);
}
// B: pre-swizzled fragment layout. For tile (kc, nt) lane L holds 16
// contiguous halves at ((kc*nt_n + nt)*32 + L)*16 -> one 32-byte load.
__device__ __forceinline__ v16h wmma_load_b_swz(const _Float16* Wsw, int nt_n,
                                                int nt, int kc, int lane) {
  return *(const v16h*)(Wsw + (((size_t)kc * nt_n + nt) * 32 + lane) * 16);
}

// One MLP layer: Xout[r,c] = relu(sum_k Xin[r,k]*W[k,c] + bias[c]).
// Rows = NSrows (mult of 16), Cin mult of 32, Cout mult of 16.
// Tiles are distributed wave-uniformly; EXEC stays all-ones around WMMA.
__device__ __forceinline__ void mlp_layer_dev(const _Float16* Xin, int Cin,
                                              const _Float16* Wsw, const float* bias,
                                              _Float16* Xout, int Cout, int NSrows,
                                              int tid, int nthreads) {
  const int lane = tid & 31, wave = tid >> 5, nw = nthreads >> 5;
  const int mt_n = NSrows >> 4, nt_n = Cout >> 4, kt_n = Cin >> 5;
  for (int t = wave; t < mt_n * nt_n; t += nw) {
    const int mt = t / nt_n, nt = t - mt * nt_n;
    v8f acc = {0.f, 0.f, 0.f, 0.f, 0.f, 0.f, 0.f, 0.f};
    for (int kc = 0; kc < kt_n; ++kc) {
      v16h a = wmma_load_a(Xin, Cin, mt * 16, kc * 32, lane);
      v16h b = wmma_load_b_swz(Wsw, nt_n, nt, kc, lane);
      acc = __builtin_amdgcn_wmma_f32_16x16x32_f16(false, a, false, b,
                                                   (short)0, acc, false, false);
    }
    const int col = nt * 16 + (lane & 15);
    const float bv = bias[col];
    const int r0 = mt * 16 + ((lane & 16) ? 8 : 0);
#pragma unroll
    for (int v = 0; v < 8; ++v) {
      float y = acc[v] + bv;
      y = fmaxf(y, 0.f);
      Xout[(r0 + v) * Cout + col] = (_Float16)y;
    }
  }
}

// ---- weight prep: fold BN scale/shift, emit WMMA-B-fragment-swizzled f16 ---
// dst[((kc*(Cout/16) + nt)*32 + lane)*16 + j] = W'[k][n]
//   with n = nt*16 + (lane&15), k = kc*32 + ((lane&16)?8:0) + (j&7) + ((j&8)?16:0).
__global__ void prep_w(const float* __restrict__ W, const float* __restrict__ b,
                       const float* __restrict__ g, const float* __restrict__ be,
                       _Float16* __restrict__ Wsw, float* __restrict__ bout,
                       int Cin, int Cout, int Kpad) {
  const int i = blockIdx.x * blockDim.x + threadIdx.x;
  if (i < Cout) bout[i] = b[i] * g[i] + be[i];
  const int total = Kpad * Cout;
  const int nt_n = Cout >> 4;
  for (int e = i; e < total; e += gridDim.x * blockDim.x) {
    const int k = e / Cout, o = e - k * Cout;
    const int kc = k >> 5, kr = k & 31;
    const int nt = o >> 4, nr = o & 15;
    const int lane = nr + ((kr & 8) ? 16 : 0);
    const int j = (kr & 7) + ((kr & 16) ? 8 : 0);
    const size_t dst = (((size_t)kc * nt_n + nt) * 32 + lane) * 16 + j;
    Wsw[dst] = (k < Cin) ? (_Float16)(W[o * Cin + k] * g[o]) : (_Float16)0.f;
  }
}

// ---- farthest point sampling (block per batch, 256 threads) ----------------
__global__ __launch_bounds__(256) void fps_kernel(const float* __restrict__ pts,
                                                  int Nn, int np, int* __restrict__ fidx) {
  const int b = blockIdx.x, tid = threadIdx.x;
  const float* base = pts + (size_t)b * Nn * 3;
  float dist[4], px[4], py[4], pz[4];
#pragma unroll
  for (int t = 0; t < 4; ++t) {
    const int n = tid + t * 256;
    if (n < Nn) { px[t] = base[n*3]; py[t] = base[n*3+1]; pz[t] = base[n*3+2]; dist[t] = 1e10f; }
    else dist[t] = -1.f;
  }
  __shared__ float sd[256];
  __shared__ int   si[256];
  __shared__ int   sfar;
  if (tid == 0) sfar = 0;
  __syncthreads();
  for (int i = 0; i < np; ++i) {
    const int far = sfar;
    if (tid == 0) fidx[b * np + i] = far;
    const float cx = base[far*3], cy = base[far*3+1], cz = base[far*3+2];
    float best = -1.f; int bidx = 0x7fffffff;
#pragma unroll
    for (int t = 0; t < 4; ++t) {
      const int n = tid + t * 256;
      if (n < Nn) {
        const float dx = px[t]-cx, dy = py[t]-cy, dz = pz[t]-cz;
        const float d2 = dx*dx + dy*dy + dz*dz;
        if (d2 < dist[t]) dist[t] = d2;
        if (dist[t] > best) { best = dist[t]; bidx = n; }
      }
    }
    __syncthreads();
    sd[tid] = best; si[tid] = bidx;
    __syncthreads();
    for (int s = 128; s > 0; s >>= 1) {
      if (tid < s) {
        const float d2 = sd[tid+s]; const int i2 = si[tid+s];
        if (d2 > sd[tid] || (d2 == sd[tid] && i2 < si[tid])) { sd[tid] = d2; si[tid] = i2; }
      }
      __syncthreads();
    }
    if (tid == 0) sfar = si[0];
    __syncthreads();
  }
}

// ---- gather sampled centers ------------------------------------------------
__global__ void gather_xyz(const float* __restrict__ pts, const int* __restrict__ fidx,
                           float* __restrict__ out, int Nn, int S, int total) {
  const int i = blockIdx.x * blockDim.x + threadIdx.x;
  if (i >= total) return;
  const int b = i / S, idx = fidx[i];
  const float* p = pts + ((size_t)b * Nn + idx) * 3;
  out[i*3+0] = p[0]; out[i*3+1] = p[1]; out[i*3+2] = p[2];
}

// ---- ball query (first nsample in-radius, ascending order, pad with first) -
__global__ void ballquery_kernel(const float* __restrict__ pts, const float* __restrict__ cen,
                                 int Nn, int S, float r2, int ns, int* __restrict__ gidx,
                                 int total) {
  const int i = blockIdx.x * blockDim.x + threadIdx.x;
  if (i >= total) return;
  const int b = i / S;
  const float cx = cen[i*3], cy = cen[i*3+1], cz = cen[i*3+2];
  const float* base = pts + (size_t)b * Nn * 3;
  int cnt = 0, first = -1;
  for (int n = 0; n < Nn && cnt < ns; ++n) {
    const float dx = base[n*3]-cx, dy = base[n*3+1]-cy, dz = base[n*3+2]-cz;
    if (dx*dx + dy*dy + dz*dz <= r2) {
      if (first < 0) first = n;
      gidx[(long long)i * ns + cnt++] = n;
    }
  }
  const int pad = (first < 0) ? 0 : first;
  for (; cnt < ns; ++cnt) gidx[(long long)i * ns + cnt] = pad;
}

// ---- SA1 fused: gather(9ch pad32) -> 32->64->64->128 -> maxpool(64) --------
__global__ __launch_bounds__(128) void sa1_fused(
    const float* __restrict__ xyz, const int* __restrict__ gidx,
    const float* __restrict__ new_xyz, const float* __restrict__ zlat,
    const _Float16* __restrict__ W0, const float* __restrict__ b0,
    const _Float16* __restrict__ W1, const float* __restrict__ b1,
    const _Float16* __restrict__ W2, const float* __restrict__ b2,
    _Float16* __restrict__ fout) {
  constexpr int NS = 64, C0 = 32, C1 = 64, C2 = 64, C3 = 128, Nn = 1024;
  extern __shared__ char smem[];
  _Float16* bufA = (_Float16*)smem;            // 64 * max(32,64) halves
  _Float16* bufB = bufA + NS * C2;             // 64 * max(64,128) halves
  const int cenI = blockIdx.x, b = cenI >> 7;
  const int tid = threadIdx.x, nth = blockDim.x;
  const float cx = new_xyz[cenI*3], cy = new_xyz[cenI*3+1], cz = new_xyz[cenI*3+2];
  const float z0 = zlat[b*2], z1 = zlat[b*2+1];
  for (int k = tid; k < NS; k += nth) {
    const int idx = gidx[(long long)cenI * NS + k];
    const float* p = xyz + ((size_t)b * Nn + idx) * 3;
    const float px = p[0], py = p[1], pz = p[2];
    _Float16* row = bufA + k * C0;
    row[0] = (_Float16)(px-cx); row[1] = (_Float16)(py-cy); row[2] = (_Float16)(pz-cz);
    row[3] = (_Float16)px;      row[4] = (_Float16)py;      row[5] = (_Float16)pz;
    row[6] = (_Float16)z0;      row[7] = (_Float16)z1;
    row[8] = (_Float16)((idx == Nn-1) ? 1.f : 0.f);
#pragma unroll
    for (int c = 9; c < C0; ++c) row[c] = (_Float16)0.f;
  }
  __syncthreads();
  mlp_layer_dev(bufA, C0, W0, b0, bufB, C1, NS, tid, nth);
  __syncthreads();
  mlp_layer_dev(bufB, C1, W1, b1, bufA, C2, NS, tid, nth);
  __syncthreads();
  mlp_layer_dev(bufA, C2, W2, b2, bufB, C3, NS, tid, nth);
  __syncthreads();
  for (int c = tid; c < C3; c += nth) {
    float m = -1e30f;
    for (int k = 0; k < NS; ++k) m = fmaxf(m, (float)bufB[k * C3 + c]);
    fout[(long long)cenI * C3 + c] = (_Float16)m;
  }
}

// ---- SA2 fused: 2 passes of 64 rows; gather(131 pad160)->128->128->256 -----
__global__ __launch_bounds__(256) void sa2_fused(
    const float* __restrict__ xyz1, const int* __restrict__ gidx,
    const float* __restrict__ new_xyz, const _Float16* __restrict__ f1,
    const _Float16* __restrict__ W0, const float* __restrict__ b0,
    const _Float16* __restrict__ W1, const float* __restrict__ b1,
    const _Float16* __restrict__ W2, const float* __restrict__ b2,
    _Float16* __restrict__ fout) {
  constexpr int NS = 128, R = 64, C0 = 160, CR = 131, C1 = 128, C2 = 128, C3 = 256;
  constexpr int Nn = 128, F = 128;
  extern __shared__ char smem[];
  _Float16* bufA = (_Float16*)smem;      // 64*160
  _Float16* bufB = bufA + R * C0;        // 64*256
  float*    mbuf = (float*)(bufB + R * C3);
  const int cenI = blockIdx.x, b = cenI >> 5;
  const int tid = threadIdx.x, nth = blockDim.x;
  const float cx = new_xyz[cenI*3], cy = new_xyz[cenI*3+1], cz = new_xyz[cenI*3+2];
  for (int c = tid; c < C3; c += nth) mbuf[c] = -1e30f;
  for (int pass = 0; pass < NS / R; ++pass) {
    __syncthreads();
    for (int e = tid; e < R * C0; e += nth) {
      const int k = e / C0, c = e - k * C0;
      const int idx = gidx[(long long)cenI * NS + pass * R + k];
      float v;
      if (c < 3) {
        const float* p = xyz1 + ((size_t)b * Nn + idx) * 3;
        v = p[c] - (c == 0 ? cx : (c == 1 ? cy : cz));
      } else if (c < CR) {
        v = (float)f1[((size_t)b * Nn + idx) * F + (c - 3)];
      } else v = 0.f;
      bufA[e] = (_Float16)v;
    }
    __syncthreads();
    mlp_layer_dev(bufA, C0, W0, b0, bufB, C1, R, tid, nth);
    __syncthreads();
    mlp_layer_dev(bufB, C1, W1, b1, bufA, C2, R, tid, nth);
    __syncthreads();
    mlp_layer_dev(bufA, C2, W2, b2, bufB, C3, R, tid, nth);
    __syncthreads();
    for (int c = tid; c < C3; c += nth) {
      float m = mbuf[c];
      for (int k = 0; k < R; ++k) m = fmaxf(m, (float)bufB[k * C3 + c]);
      mbuf[c] = m;
    }
  }
  __syncthreads();
  for (int c = tid; c < C3; c += nth) fout[(long long)cenI * C3 + c] = (_Float16)mbuf[c];
}

// ---- SA3 fused (group-all, block per batch): 259 pad288 ->256->256->512 ----
__global__ __launch_bounds__(256) void sa3_fused(
    const float* __restrict__ xyz2, const _Float16* __restrict__ f2,
    const _Float16* __restrict__ W0, const float* __restrict__ b0,
    const _Float16* __restrict__ W1, const float* __restrict__ b1,
    const _Float16* __restrict__ W2, const float* __restrict__ b2,
    _Float16* __restrict__ xout) {
  constexpr int R = 32, C0 = 288, CR = 259, C1 = 256, C2 = 256, C3 = 512, F = 256;
  extern __shared__ char smem[];
  _Float16* bufA = (_Float16*)smem;     // 32*288
  _Float16* bufB = bufA + R * C0;       // 32*512
  const int b = blockIdx.x, tid = threadIdx.x, nth = blockDim.x;
  for (int e = tid; e < R * C0; e += nth) {
    const int s = e / C0, c = e - s * C0;
    float v;
    if (c < 3)        v = xyz2[((size_t)b * R + s) * 3 + c];
    else if (c < CR)  v = (float)f2[((size_t)b * R + s) * F + (c - 3)];
    else              v = 0.f;
    bufA[e] = (_Float16)v;
  }
  __syncthreads();
  mlp_layer_dev(bufA, C0, W0, b0, bufB, C1, R, tid, nth);
  __syncthreads();
  mlp_layer_dev(bufB, C1, W1, b1, bufA, C2, R, tid, nth);
  __syncthreads();
  mlp_layer_dev(bufA, C2, W2, b2, bufB, C3, R, tid, nth);
  __syncthreads();
  for (int c = tid; c < C3; c += nth) {
    float m = -1e30f;
    for (int k = 0; k < R; ++k) m = fmaxf(m, (float)bufB[k * C3 + c]);
    xout[(long long)b * C3 + c] = (_Float16)m;
  }
}

// ---- generic WMMA GEMM + bias + relu (for the FC layers) -------------------
__global__ __launch_bounds__(128) void gemm_relu(const _Float16* __restrict__ A,
                                                 const _Float16* __restrict__ Wsw,
                                                 const float* __restrict__ bias,
                                                 _Float16* __restrict__ C,
                                                 int M, int K, int N) {
  const int wave = threadIdx.x >> 5, lane = threadIdx.x & 31;
  const int tile = blockIdx.x * (blockDim.x >> 5) + wave;
  const int nt_n = N >> 4, mt_n = M >> 4;
  if (tile >= mt_n * nt_n) return;
  const int mt = tile / nt_n, nt = tile - mt * nt_n;
  v8f acc = {0.f, 0.f, 0.f, 0.f, 0.f, 0.f, 0.f, 0.f};
  for (int kc = 0; kc < (K >> 5); ++kc) {
    v16h a = wmma_load_a(A, K, mt * 16, kc * 32, lane);
    v16h b = wmma_load_b_swz(Wsw, nt_n, nt, kc, lane);
    acc = __builtin_amdgcn_wmma_f32_16x16x32_f16(false, a, false, b,
                                                 (short)0, acc, false, false);
  }
  const int col = nt * 16 + (lane & 15);
  const float bv = bias[col];
  const int r0 = mt * 16 + ((lane & 16) ? 8 : 0);
#pragma unroll
  for (int v = 0; v < 8; ++v) {
    float y = fmaxf(acc[v] + bv, 0.f);
    C[(size_t)(r0 + v) * N + col] = (_Float16)y;
  }
}

// ---- head: q = norm(x@Wq^T + bq), t = x@Wt^T + bt --------------------------
__global__ __launch_bounds__(64) void head_kernel(const _Float16* __restrict__ x,
                                                  const float* __restrict__ Wq,
                                                  const float* __restrict__ bq,
                                                  const float* __restrict__ Wt,
                                                  const float* __restrict__ bt,
                                                  float* __restrict__ out) {
  __shared__ float red[7][64];
  const int b = blockIdx.x, tid = threadIdx.x;
  float acc[7] = {0.f, 0.f, 0.f, 0.f, 0.f, 0.f, 0.f};
  for (int i = tid; i < 1024; i += 64) {
    const float xv = (float)x[(size_t)b * 1024 + i];
#pragma unroll
    for (int j = 0; j < 4; ++j) acc[j]     += xv * Wq[j * 1024 + i];
#pragma unroll
    for (int j = 0; j < 3; ++j) acc[4 + j] += xv * Wt[j * 1024 + i];
  }
#pragma unroll
  for (int j = 0; j < 7; ++j) red[j][tid] = acc[j];
  __syncthreads();
  if (tid == 0) {
    float s[7];
#pragma unroll
    for (int j = 0; j < 7; ++j) {
      float v = 0.f;
      for (int k = 0; k < 64; ++k) v += red[j][k];
      s[j] = v;
    }
    float q0 = s[0] + bq[0], q1 = s[1] + bq[1], q2 = s[2] + bq[2], q3 = s[3] + bq[3];
    const float nrm = fmaxf(sqrtf(q0*q0 + q1*q1 + q2*q2 + q3*q3), 1e-12f);
    out[b*7+0] = q0/nrm; out[b*7+1] = q1/nrm; out[b*7+2] = q2/nrm; out[b*7+3] = q3/nrm;
    out[b*7+4] = s[4] + bt[0]; out[b*7+5] = s[5] + bt[1]; out[b*7+6] = s[6] + bt[2];
  }
}

// ---------------------------------------------------------------------------
static int find_sz(const int* s, int n, int v) {
  for (int i = 0; i < n; ++i) if (s[i] == v) return i;
  return -1;
}

extern "C" void kernel_launch(void* const* d_in, const int* in_sizes, int n_in,
                              void* d_out, int out_size, void* d_ws, size_t ws_size,
                              hipStream_t stream) {
  constexpr int B = 128, N = 1024;
  // ---- locate inputs by unique sizes (robust to pytree flatten order) ----
  const int i_xyz = find_sz(in_sizes, n_in, B * N * 3);
  const int i_sa1 = find_sz(in_sizes, n_in, 64 * 9);
  const int i_sa2 = find_sz(in_sizes, n_in, 128 * 131);
  const int i_sa3 = find_sz(in_sizes, n_in, 256 * 259);
  const int i_fc  = find_sz(in_sizes, n_in, 1024 * 512);
  int i_q = -1, i_t = -1;
  for (int i = 0; i + 1 < n_in; ++i) {
    if (in_sizes[i] == 4096 && in_sizes[i + 1] == 4) { i_q = i; break; }
  }
  for (int i = 0; i + 1 < n_in; ++i) {
    if (in_sizes[i] == 3072 && in_sizes[i + 1] == 3) { i_t = i; break; }
  }
  int i_z = -1;
  for (int i = 0; i < n_in; ++i) {
    if (in_sizes[i] != 256) continue;
    const bool inblk = (i >= i_sa1 && i < i_sa1 + 12) || (i >= i_sa2 && i < i_sa2 + 12) ||
                       (i >= i_sa3 && i < i_sa3 + 12) || (i >= i_fc && i < i_fc + 8) ||
                       i == i_q || i == i_q + 1 || i == i_t || i == i_t + 1 || i == i_xyz;
    if (!inblk) { i_z = i; break; }
  }
  if (i_xyz < 0 || i_sa1 < 0 || i_sa2 < 0 || i_sa3 < 0 || i_fc < 0 ||
      i_q < 0 || i_t < 0 || i_z < 0) return;

  const float* xyz  = (const float*)d_in[i_xyz];
  const float* zlat = (const float*)d_in[i_z];
  const float *Wq = (const float*)d_in[i_q], *bq = (const float*)d_in[i_q + 1];
  const float *Wtr = (const float*)d_in[i_t], *btr = (const float*)d_in[i_t + 1];

  // raw layer params: (W,b,g,be) x3 for each SA, x2 for FC
  const float *raw[11][4];
  for (int l = 0; l < 3; ++l)
    for (int j = 0; j < 4; ++j) raw[l][j] = (const float*)d_in[i_sa1 + l * 4 + j];
  for (int l = 0; l < 3; ++l)
    for (int j = 0; j < 4; ++j) raw[3 + l][j] = (const float*)d_in[i_sa2 + l * 4 + j];
  for (int l = 0; l < 3; ++l)
    for (int j = 0; j < 4; ++j) raw[6 + l][j] = (const float*)d_in[i_sa3 + l * 4 + j];
  for (int l = 0; l < 2; ++l)
    for (int j = 0; j < 4; ++j) raw[9 + l][j] = (const float*)d_in[i_fc + l * 4 + j];

  // layer dims: {Cin, Cout, Kpad}
  static const int LD[11][3] = {
      {9, 64, 32},    {64, 64, 64},   {64, 128, 64},     // sa1
      {131, 128, 160},{128, 128, 128},{128, 256, 128},   // sa2
      {259, 256, 288},{256, 256, 256},{256, 512, 256},   // sa3
      {512, 1024, 512},{1024, 1024, 1024}};              // fc

  // ---- workspace bump allocator ----
  char* ws = (char*)d_ws;
  size_t off = 0;
  auto alloc = [&](size_t bytes) -> void* {
    void* p = ws + off;
    off = (off + bytes + 255) & ~(size_t)255;
    return p;
  };
  int*   fidx1 = (int*)alloc((size_t)B * 128 * 4);
  float* xyz1  = (float*)alloc((size_t)B * 128 * 3 * 4);
  int*   gidx1 = (int*)alloc((size_t)B * 128 * 64 * 4);
  _Float16* f1 = (_Float16*)alloc((size_t)B * 128 * 128 * 2);
  int*   fidx2 = (int*)alloc((size_t)B * 32 * 4);
  float* xyz2  = (float*)alloc((size_t)B * 32 * 3 * 4);
  int*   gidx2 = (int*)alloc((size_t)B * 32 * 128 * 4);
  _Float16* f2 = (_Float16*)alloc((size_t)B * 32 * 256 * 2);
  _Float16* x512 = (_Float16*)alloc((size_t)B * 512 * 2);
  _Float16* fcb1 = (_Float16*)alloc((size_t)B * 1024 * 2);
  _Float16* fcb2 = (_Float16*)alloc((size_t)B * 1024 * 2);
  _Float16* Wt[11];
  float*    Bi[11];
  for (int l = 0; l < 11; ++l) {
    Wt[l] = (_Float16*)alloc((size_t)LD[l][2] * LD[l][1] * 2);
    Bi[l] = (float*)alloc((size_t)LD[l][1] * 4);
  }

  // ---- prep weights (fold BN, pad K, swizzle to WMMA B-fragment order) ----
  for (int l = 0; l < 11; ++l) {
    const int total = LD[l][2] * LD[l][1];
    const int grid = (total + 255) / 256;
    prep_w<<<grid, 256, 0, stream>>>(raw[l][0], raw[l][1], raw[l][2], raw[l][3],
                                     Wt[l], Bi[l], LD[l][0], LD[l][1], LD[l][2]);
  }

  // ---- SA1 ----
  fps_kernel<<<B, 256, 0, stream>>>(xyz, N, 128, fidx1);
  gather_xyz<<<(B * 128 + 255) / 256, 256, 0, stream>>>(xyz, fidx1, xyz1, N, 128, B * 128);
  ballquery_kernel<<<(B * 128 + 255) / 256, 256, 0, stream>>>(
      xyz, xyz1, N, 128, 0.02f * 0.02f, 64, gidx1, B * 128);
  {
    const size_t sm = (size_t)(64 * 64 + 64 * 128) * sizeof(_Float16);  // 24 KB
    sa1_fused<<<B * 128, 128, sm, stream>>>(xyz, gidx1, xyz1, zlat,
                                            Wt[0], Bi[0], Wt[1], Bi[1], Wt[2], Bi[2], f1);
  }
  // ---- SA2 ----
  fps_kernel<<<B, 256, 0, stream>>>(xyz1, 128, 32, fidx2);
  gather_xyz<<<(B * 32 + 255) / 256, 256, 0, stream>>>(xyz1, fidx2, xyz2, 128, 32, B * 32);
  ballquery_kernel<<<(B * 32 + 255) / 256, 256, 0, stream>>>(
      xyz1, xyz2, 128, 32, 0.04f * 0.04f, 128, gidx2, B * 32);
  {
    const size_t sm = (size_t)(64 * 160 + 64 * 256) * sizeof(_Float16) + 256 * sizeof(float);
    sa2_fused<<<B * 32, 256, sm, stream>>>(xyz1, gidx2, xyz2, f1,
                                           Wt[3], Bi[3], Wt[4], Bi[4], Wt[5], Bi[5], f2);
  }
  // ---- SA3 (group-all) ----
  {
    const size_t sm = (size_t)(32 * 288 + 32 * 512) * sizeof(_Float16);  // 50 KB
    sa3_fused<<<B, 256, sm, stream>>>(xyz2, f2,
                                      Wt[6], Bi[6], Wt[7], Bi[7], Wt[8], Bi[8], x512);
  }
  // ---- FC ----
  {
    const int tiles0 = (B / 16) * (1024 / 16);  // 512 tiles, 4 waves/block
    gemm_relu<<<(tiles0 + 3) / 4, 128, 0, stream>>>(x512, Wt[9], Bi[9], fcb1, B, 512, 1024);
    gemm_relu<<<(tiles0 + 3) / 4, 128, 0, stream>>>(fcb1, Wt[10], Bi[10], fcb2, B, 1024, 1024);
  }
  // ---- head ----
  head_kernel<<<B, 64, 0, stream>>>(fcb2, Wq, bq, Wtr, btr, (float*)d_out);
}
